// AttnBlock_27006754357374
// MI455X (gfx1250) — matrix operand order repaired and checked by reading, not compiled, not run on hardware
//
#include <hip/hip_runtime.h>
#include <hip/hip_bf16.h>
#include <math.h>

typedef __bf16 bf16;
typedef __attribute__((ext_vector_type(16))) __bf16 bf16x16;
typedef __attribute__((ext_vector_type(8)))  __bf16 bf16x8;
typedef __attribute__((ext_vector_type(8)))  float  v8f;

#define B_ 2
#define C_ 256
#define H_ 64
#define W_ 64
#define HW_ 4096
#define GROUPS_ 32
#define WF_ 33            // rfft width bins
#define FC_ (H_ * WF_)    // 2112 freq-domain pixels per channel

static __device__ inline bf16x16 cat8(bf16x8 lo, bf16x8 hi) {
  return __builtin_shufflevector(lo, hi, 0,1,2,3,4,5,6,7,8,9,10,11,12,13,14,15);
}

// Async DMA: copy 16 bytes global -> LDS per lane (ASYNCcnt-tracked, no VGPR data).
static __device__ inline void async_ld_b128(unsigned lds_off, const void* gptr) {
  asm volatile("global_load_async_to_lds_b128 %0, %1, off"
               :: "v"(lds_off), "v"(gptr) : "memory");
}
static __device__ inline void wait_asynccnt0() {
  asm volatile("s_wait_asynccnt 0x0" ::: "memory");
}
static __device__ inline unsigned lds_addr_of(const void* p) {
  return (unsigned)(unsigned long long)p;   // LDS flat addr low 32 bits == LDS offset
}

// ---------------------------------------------------------------------------
// Batched bf16 GEMM, f32 accumulate via v_wmma_f32_16x16x32_bf16.
// C[z] = alpha * opA(A[z]) * opB(B[z]) (+ bias[row]) (+ add[z])
// opA(A)[m,k] = TA ? A[k*lda+m] : A[m*lda+k];  opB(B)[k,n] = TB ? B[n*ldb+k] : B[k*ldb+n]
// Block: 128 threads (4 waves), 64x64 output tile, K stepped by 32.
// Contiguous-k staging directions use global_load_async_to_lds_b128.
// All lda/ldb/offsets are multiples of 8 elems -> 16B-aligned vector accesses.
// ---------------------------------------------------------------------------
template <int TA, int TB>
__global__ __launch_bounds__(128)
void gemm_bf16_kernel(const bf16* __restrict__ A, const bf16* __restrict__ Bm,
                      float* __restrict__ C,
                      int M, int N, int K, int lda, int ldb, int ldc,
                      float alpha,
                      const float* __restrict__ bias,
                      const float* __restrict__ add,
                      long long sA, long long sB, long long sC) {
  (void)M; (void)N;
  __shared__ alignas(16) bf16 As[64 * 40];
  __shared__ alignas(16) bf16 Bs[64 * 40];
  const int tid  = threadIdx.x;
  const int wave = tid >> 5;
  const int lane = tid & 31;
  const int half = lane >> 4;
  const int l16  = lane & 15;
  const int m0 = blockIdx.y * 64;
  const int n0 = blockIdx.x * 64;
  const bf16* Ab = A + sA * (long long)blockIdx.z;
  const bf16* Bb = Bm + sB * (long long)blockIdx.z;
  float* Cb = C + sC * (long long)blockIdx.z;

  v8f acc[4];
  for (int t = 0; t < 4; t++)
    for (int r = 0; r < 8; r++) acc[t][r] = 0.0f;

  for (int kt = 0; kt < K; kt += 32) {
    // ---- stage A tile (64 rows x 32 k) into LDS As[m][k], k-contiguous ----
    if (TA == 0) {
      // global k-contiguous, LDS contiguous: async DMA, 16B per lane
#pragma unroll
      for (int i = 0; i < 2; i++) {
        int chunk = tid + i * 128;            // 256 chunks of 8 bf16
        int m = chunk >> 2, c4 = chunk & 3;
        const bf16* src = Ab + (long long)(m0 + m) * lda + (kt + c4 * 8);
        async_ld_b128(lds_addr_of(&As[m * 40 + c4 * 8]), src);
      }
    } else {
      // global m-contiguous: wide load, 8-way LDS scatter
#pragma unroll
      for (int i = 0; i < 2; i++) {
        int chunk = tid + i * 128;            // 32 k x 8 m-groups
        int k = chunk >> 3, m8 = (chunk & 7) * 8;
        bf16x8 v = *(const bf16x8*)(Ab + (long long)(kt + k) * lda + (m0 + m8));
#pragma unroll
        for (int e = 0; e < 8; e++) As[(m8 + e) * 40 + k] = v[e];
      }
    }
    // ---- stage B tile into LDS Bs[n][k], k-contiguous ----
    if (TB == 1) {
      // global k-contiguous, LDS contiguous: async DMA
#pragma unroll
      for (int i = 0; i < 2; i++) {
        int chunk = tid + i * 128;
        int n = chunk >> 2, c4 = chunk & 3;
        const bf16* src = Bb + (long long)(n0 + n) * ldb + (kt + c4 * 8);
        async_ld_b128(lds_addr_of(&Bs[n * 40 + c4 * 8]), src);
      }
    } else {
      // global n-contiguous: wide load, 8-way LDS scatter
#pragma unroll
      for (int i = 0; i < 2; i++) {
        int chunk = tid + i * 128;
        int k = chunk >> 3, n8 = (chunk & 7) * 8;
        bf16x8 v = *(const bf16x8*)(Bb + (long long)(kt + k) * ldb + (n0 + n8));
#pragma unroll
        for (int e = 0; e < 8; e++) Bs[(n8 + e) * 40 + k] = v[e];
      }
    }
    if (TA == 0 || TB == 1) wait_asynccnt0();
    __syncthreads();

    // A fragment: lane row = wave*16 + l16; elems 0-7 -> k=half*8.., 8-15 -> k=16+half*8..
    const int arow = (wave * 16 + l16) * 40;
    bf16x8 a_lo = *(const bf16x8*)&As[arow + half * 8];
    bf16x8 a_hi = *(const bf16x8*)&As[arow + 16 + half * 8];
    bf16x16 afrag = cat8(a_lo, a_hi);

#pragma unroll
    for (int nt = 0; nt < 4; nt++) {
      // B fragment: lane col = nt*16 + l16; elems 0-7 -> k=half*8.., 8-15 -> k=16+half*8..
      const int brow = (nt * 16 + l16) * 40;
      bf16x8 b_lo = *(const bf16x8*)&Bs[brow + half * 8];
      bf16x8 b_hi = *(const bf16x8*)&Bs[brow + 16 + half * 8];
      bf16x16 bfrag = cat8(b_lo, b_hi);
      acc[nt] = __builtin_amdgcn_wmma_f32_16x16x32_bf16(
          false, afrag, false, bfrag, (short)0, acc[nt], false, false);
    }
    __syncthreads();
  }

  // D layout: VGPR r -> row m = r + 8*half, col n = l16 (within 16-tile)
#pragma unroll
  for (int nt = 0; nt < 4; nt++) {
    int col = n0 + nt * 16 + l16;
#pragma unroll
    for (int r = 0; r < 8; r++) {
      int row = m0 + wave * 16 + r + 8 * half;
      float v = alpha * acc[nt][r];
      if (bias) v += bias[row];
      long long off = (long long)row * ldc + col;
      if (add) v += add[sC * (long long)blockIdx.z + off];
      Cb[off] = v;
    }
  }
}

// ---------------------------------------------------------------------------
__global__ void f32_to_bf16_kernel(const float* __restrict__ s, bf16* __restrict__ d, long long n) {
  long long i = (long long)blockIdx.x * blockDim.x + threadIdx.x;
  if (i < n) d[i] = (bf16)s[i];
}

__global__ void subtract_kernel(const float* __restrict__ a, const float* __restrict__ b,
                                float* __restrict__ d, long long n) {
  long long i = (long long)blockIdx.x * blockDim.x + threadIdx.x;
  if (i < n) d[i] = a[i] - b[i];
}

// ---------------------------------------------------------------------------
__global__ __launch_bounds__(256)
void groupnorm_kernel(const float* __restrict__ x, const float* __restrict__ scale,
                      const float* __restrict__ bias, float* __restrict__ out) {
  const int b = blockIdx.x / GROUPS_, g = blockIdx.x % GROUPS_;
  const int CPG = C_ / GROUPS_;                 // 8
  const long long base = ((long long)b * C_ + (long long)g * CPG) * HW_;
  const float* xp = x + base;
  float* op = out + base;
  const int NG = CPG * HW_;                     // 32768
  __shared__ float r1[256], r2[256];
  float s = 0.f, ss = 0.f;
  for (int i = threadIdx.x; i < NG; i += 256) { float v = xp[i]; s += v; ss += v * v; }
  r1[threadIdx.x] = s; r2[threadIdx.x] = ss; __syncthreads();
  for (int st = 128; st > 0; st >>= 1) {
    if (threadIdx.x < st) { r1[threadIdx.x] += r1[threadIdx.x + st]; r2[threadIdx.x] += r2[threadIdx.x + st]; }
    __syncthreads();
  }
  float mean = r1[0] / NG;
  float var  = r2[0] / NG - mean * mean;
  float inv  = rsqrtf(var + 1e-6f);
  for (int i = threadIdx.x; i < NG; i += 256) {
    int c = g * CPG + i / HW_;
    op[i] = (xp[i] - mean) * inv * scale[c] + bias[c];
  }
}

// ---------------------------------------------------------------------------
// Depthwise conv, SAME padding, weights (C,1,ks,ks) f32. Grid = B*C blocks.
__global__ __launch_bounds__(256)
void dwconv_kernel(const float* __restrict__ src, const float* __restrict__ w,
                   const float* __restrict__ bias, float* __restrict__ dst,
                   int C, int H, int W, int ks) {
  const int bc = blockIdx.x, c = bc % C;
  const long long base = (long long)bc * H * W;
  const float* sp = src + base;
  float* dp = dst + base;
  const float* wp = w + (long long)c * ks * ks;
  const float bv = bias ? bias[c] : 0.f;
  const int pad = ks / 2;
  for (int p = threadIdx.x; p < H * W; p += 256) {
    int y = p / W, xx = p % W;
    float acc = bv;
    for (int ky = 0; ky < ks; ky++) {
      int yy = y + ky - pad;
      if (yy < 0 || yy >= H) continue;
      for (int kx = 0; kx < ks; kx++) {
        int xc = xx + kx - pad;
        if (xc < 0 || xc >= W) continue;
        acc += wp[ky * ks + kx] * sp[yy * W + xc];
      }
    }
    dp[p] = acc;
  }
}

// Depthwise Gaussian blur, weights synthesized on the fly. Grid = B*C blocks.
__global__ __launch_bounds__(256)
void gauss_dw_kernel(const float* __restrict__ src, float* __restrict__ dst,
                     int H, int W, int ks, float sigma) {
  __shared__ float wk[49];
  if (threadIdx.x == 0) {
    float g[7]; float s = 0.f;
    for (int i = 0; i < ks; i++) {
      float t = (float)i - (ks - 1) * 0.5f;
      g[i] = expf(-(t * t) / (2.f * sigma * sigma));
      s += g[i];
    }
    for (int i = 0; i < ks; i++) g[i] /= s;
    for (int y = 0; y < ks; y++)
      for (int x = 0; x < ks; x++) wk[y * ks + x] = g[y] * g[x];
  }
  __syncthreads();
  const long long base = (long long)blockIdx.x * H * W;
  const float* sp = src + base;
  float* dp = dst + base;
  const int pad = ks / 2;
  for (int p = threadIdx.x; p < H * W; p += 256) {
    int y = p / W, xx = p % W;
    float acc = 0.f;
    for (int ky = 0; ky < ks; ky++) {
      int yy = y + ky - pad;
      if (yy < 0 || yy >= H) continue;
      for (int kx = 0; kx < ks; kx++) {
        int xc = xx + kx - pad;
        if (xc < 0 || xc >= W) continue;
        acc += wk[ky * ks + kx] * sp[yy * W + xc];
      }
    }
    dp[p] = acc;
  }
}

// ---------------------------------------------------------------------------
// Row softmax: one block per row. src/dst may alias.
__global__ __launch_bounds__(256)
void softmax_rows_kernel(const float* __restrict__ src, float* __restrict__ dst, int ncols) {
  const long long row = blockIdx.x;
  const float* sp = src + row * (long long)ncols;
  float* dp = dst + row * (long long)ncols;
  __shared__ float red[256];
  float m = -3.4e38f;
  for (int i = threadIdx.x; i < ncols; i += 256) m = fmaxf(m, sp[i]);
  red[threadIdx.x] = m; __syncthreads();
  for (int st = 128; st > 0; st >>= 1) {
    if (threadIdx.x < st) red[threadIdx.x] = fmaxf(red[threadIdx.x], red[threadIdx.x + st]);
    __syncthreads();
  }
  const float mx = red[0]; __syncthreads();
  float s = 0.f;
  for (int i = threadIdx.x; i < ncols; i += 256) s += expf(sp[i] - mx);
  red[threadIdx.x] = s; __syncthreads();
  for (int st = 128; st > 0; st >>= 1) {
    if (threadIdx.x < st) red[threadIdx.x] += red[threadIdx.x + st];
    __syncthreads();
  }
  const float inv = 1.f / red[0];
  for (int i = threadIdx.x; i < ncols; i += 256) dp[i] = expf(sp[i] - mx) * inv;
}

// ---------------------------------------------------------------------------
// Naive DFT kernels for 64-point rfft2/irfft2 (tiny total work).
__global__ void rfft_w_kernel(const float* __restrict__ src, float* __restrict__ dre,
                              float* __restrict__ dim, long long rows) {
  long long t = (long long)blockIdx.x * blockDim.x + threadIdx.x;
  if (t >= rows * WF_) return;
  int k = (int)(t % WF_);
  long long r = t / WF_;
  const float* sp = src + r * W_;
  float re = 0.f, im = 0.f;
  const float w0 = -2.f * 3.14159265358979323846f * (float)k / (float)W_;
  for (int w = 0; w < W_; w++) {
    float s, c; __sincosf(w0 * (float)w, &s, &c);
    re += sp[w] * c; im += sp[w] * s;
  }
  dre[r * WF_ + k] = re;
  dim[r * WF_ + k] = im;
}

// Complex DFT over H axis of (BC, H, WF_). sign=-1 fwd (scale 1), sign=+1 inv (scale 1/64).
__global__ void fft_h_kernel(const float* __restrict__ sre, const float* __restrict__ sim,
                             float* __restrict__ dre, float* __restrict__ dim,
                             long long nbc, float sign, float scale) {
  long long t = (long long)blockIdx.x * blockDim.x + threadIdx.x;
  if (t >= nbc * FC_) return;
  int kw = (int)(t % WF_);
  int kh = (int)((t / WF_) % H_);
  long long bc = t / FC_;
  const float* pr = sre + bc * FC_;
  const float* pi = sim + bc * FC_;
  float re = 0.f, im = 0.f;
  const float w0 = sign * 2.f * 3.14159265358979323846f * (float)kh / (float)H_;
  for (int h = 0; h < H_; h++) {
    float s, c; __sincosf(w0 * (float)h, &s, &c);
    float a = pr[h * WF_ + kw], b = pi[h * WF_ + kw];
    re += a * c - b * s;
    im += a * s + b * c;
  }
  dre[t] = re * scale;
  dim[t] = im * scale;
}

__global__ void irfft_w_kernel(const float* __restrict__ sre, const float* __restrict__ sim,
                               float* __restrict__ dst, long long rows) {
  long long t = (long long)blockIdx.x * blockDim.x + threadIdx.x;
  if (t >= rows * W_) return;
  int w = (int)(t % W_);
  long long r = t / W_;
  const float* pr = sre + r * WF_;
  const float* pi = sim + r * WF_;
  float acc = pr[0];
  const float w0 = 2.f * 3.14159265358979323846f * (float)w / (float)W_;
  for (int k = 1; k < 32; k++) {
    float s, c; __sincosf(w0 * (float)k, &s, &c);
    acc += 2.f * (pr[k] * c - pi[k] * s);
  }
  acc += ((w & 1) ? -1.f : 1.f) * pr[32];   // Nyquist bin
  dst[r * W_ + w] = acc * (1.f / (float)W_);
}

__global__ void amp_phase_kernel(const float* __restrict__ er, const float* __restrict__ ei,
                                 const float* __restrict__ dr, const float* __restrict__ di,
                                 float* __restrict__ amp, float* __restrict__ pha, long long n) {
  long long i = (long long)blockIdx.x * blockDim.x + threadIdx.x;
  if (i >= n) return;
  amp[i] = sqrtf(er[i] * er[i] + ei[i] * ei[i]);
  pha[i] = atan2f(di[i], dr[i]);
}

__global__ void polar_kernel(const float* __restrict__ amp, const float* __restrict__ pha,
                             float* __restrict__ pc, float* __restrict__ ps, long long n) {
  long long i = (long long)blockIdx.x * blockDim.x + threadIdx.x;
  if (i >= n) return;
  float s, c; __sincosf(pha[i], &s, &c);
  pc[i] = amp[i] * c;
  ps[i] = amp[i] * s;
}

// y = x + rec; out = 2y - mean_hw(y). One block per (b,c).
__global__ __launch_bounds__(256)
void final_kernel(const float* __restrict__ x, const float* __restrict__ rec,
                  float* __restrict__ out) {
  const long long base = (long long)blockIdx.x * HW_;
  const float* xp = x + base;
  const float* rp = rec + base;
  float* op = out + base;
  __shared__ float red[256];
  float s = 0.f;
  for (int i = threadIdx.x; i < HW_; i += 256) s += xp[i] + rp[i];
  red[threadIdx.x] = s; __syncthreads();
  for (int st = 128; st > 0; st >>= 1) {
    if (threadIdx.x < st) red[threadIdx.x] += red[threadIdx.x + st];
    __syncthreads();
  }
  const float mean = red[0] / (float)HW_;
  for (int i = threadIdx.x; i < HW_; i += 256) {
    float y = xp[i] + rp[i];
    op[i] = 2.f * y - mean;
  }
}

// ---------------------------------------------------------------------------
extern "C" void kernel_launch(void* const* d_in, const int* in_sizes, int n_in,
                              void* d_out, int out_size, void* d_ws, size_t ws_size,
                              hipStream_t stream) {
  (void)in_sizes; (void)n_in; (void)out_size; (void)ws_size;
  const float* x     = (const float*)d_in[0];
  const float* gns   = (const float*)d_in[1];
  const float* gnb   = (const float*)d_in[2];
  const float* q1w   = (const float*)d_in[3];
  const float* q1b   = (const float*)d_in[4];
  const float* q2w   = (const float*)d_in[5];
  const float* q2b   = (const float*)d_in[6];
  const float* k1w   = (const float*)d_in[7];
  const float* k1b   = (const float*)d_in[8];
  const float* k2w   = (const float*)d_in[9];
  const float* k2b   = (const float*)d_in[10];
  const float* v1w   = (const float*)d_in[11];
  const float* v1b   = (const float*)d_in[12];
  const float* v2w   = (const float*)d_in[13];
  const float* v2b   = (const float*)d_in[14];
  const float* projw = (const float*)d_in[15];
  const float* projb = (const float*)d_in[16];
  const float* midw  = (const float*)d_in[17];
  const float* midb  = (const float*)d_in[18];
  const float* postw = (const float*)d_in[19];
  const float* postb = (const float*)d_in[20];
  const float* c1w   = (const float*)d_in[21];
  const float* c1b   = (const float*)d_in[22];
  float* out = (float*)d_out;

  const long long P  = (long long)B_ * C_ * HW_;      // 2,097,152
  const long long PS = (long long)B_ * HW_ * HW_;     // 33,554,432
  const long long PF = (long long)B_ * C_ * FC_;      // 1,081,344
  const long long CC = (long long)B_ * C_ * C_;       // 131,072

  char* ws = (char*)d_ws;
  size_t off = 0;
  auto alloc = [&](size_t bytes) -> void* {
    void* p = ws + off;
    off = (off + bytes + 255) & ~(size_t)255;
    return p;
  };
  float* h_f   = (float*)alloc(P * 4);
  float* t0f   = (float*)alloc(P * 4);
  float* qf    = (float*)alloc(P * 4);
  float* kf    = (float*)alloc(P * 4);
  float* vf    = (float*)alloc(P * 4);
  float* h2f   = (float*)alloc(P * 4);
  float* h2p   = (float*)alloc(P * 4);
  float* G1f   = (float*)alloc(P * 4);
  float* G2f   = (float*)alloc(P * 4);
  float* Lf    = (float*)alloc(P * 4);
  float* smf   = (float*)alloc(P * 4);
  float* faf   = (float*)alloc(P * 4);
  float* recf  = (float*)alloc(P * 4);
  bf16* hb     = (bf16*)alloc(P * 2);   // also reused as h2 bf16
  bf16* qb     = (bf16*)alloc(P * 2);
  bf16* kb     = (bf16*)alloc(P * 2);
  bf16* vb     = (bf16*)alloc(P * 2);
  bf16* xb     = (bf16*)alloc(P * 2);   // L0 / L bf16
  bf16* smb    = (bf16*)alloc(P * 2);
  bf16* wqb    = (bf16*)alloc((size_t)C_ * C_ * 2);
  bf16* wkb    = (bf16*)alloc((size_t)C_ * C_ * 2);
  bf16* wvb    = (bf16*)alloc((size_t)C_ * C_ * 2);
  bf16* wpb    = (bf16*)alloc((size_t)C_ * C_ * 2);
  bf16* wpostb = (bf16*)alloc((size_t)C_ * C_ * 2);
  float* attf  = (float*)alloc(CC * 4);
  bf16* attb   = (bf16*)alloc(CC * 2);
  float* scores  = (float*)alloc(PS * 4);
  bf16*  scoresb = (bf16*)alloc(PS * 2);
  float* tre   = (float*)alloc(PF * 4);
  float* tim   = (float*)alloc(PF * 4);
  float* FeR   = (float*)alloc(PF * 4);
  float* FeI   = (float*)alloc(PF * 4);
  float* FdR   = (float*)alloc(PF * 4);
  float* FdI   = (float*)alloc(PF * 4);
  float* ampf  = (float*)alloc(PF * 4);
  float* phar  = (float*)alloc(PF * 4);
  float* phaf  = (float*)alloc(PF * 4);
  float* pcf   = (float*)alloc(PF * 4);
  float* psf   = (float*)alloc(PF * 4);
  float* realf = (float*)alloc(PF * 4);
  float* imagf = (float*)alloc(PF * 4);
  bf16* pcb    = (bf16*)alloc(PF * 2);

  auto cvt = [&](const float* s, bf16* d, long long n) {
    f32_to_bf16_kernel<<<dim3((unsigned)((n + 255) / 256)), 256, 0, stream>>>(s, d, n);
  };
  auto gemm = [&](const bf16* A, const bf16* Bm, float* Cp, int M, int N, int K,
                  int lda, int ldb, int ldc, int tA, int tB, float alpha,
                  const float* bias, const float* add,
                  long long sA, long long sB, long long sC) {
    dim3 g((unsigned)(N / 64), (unsigned)(M / 64), (unsigned)B_);
    if (tA == 0 && tB == 0)
      gemm_bf16_kernel<0,0><<<g, 128, 0, stream>>>(A, Bm, Cp, M, N, K, lda, ldb, ldc, alpha, bias, add, sA, sB, sC);
    else if (tA == 1 && tB == 0)
      gemm_bf16_kernel<1,0><<<g, 128, 0, stream>>>(A, Bm, Cp, M, N, K, lda, ldb, ldc, alpha, bias, add, sA, sB, sC);
    else if (tA == 0 && tB == 1)
      gemm_bf16_kernel<0,1><<<g, 128, 0, stream>>>(A, Bm, Cp, M, N, K, lda, ldb, ldc, alpha, bias, add, sA, sB, sC);
    else
      gemm_bf16_kernel<1,1><<<g, 128, 0, stream>>>(A, Bm, Cp, M, N, K, lda, ldb, ldc, alpha, bias, add, sA, sB, sC);
  };

  const long long sP = (long long)C_ * HW_;     // per-batch stride for (C,HW) fields
  const long long sS = (long long)HW_ * HW_;    // per-batch stride for scores
  const long long sF = (long long)C_ * FC_;     // per-batch stride freq fields
  const long long sA2 = (long long)C_ * C_;     // per-batch stride for att

  // 1) GroupNorm
  groupnorm_kernel<<<B_ * GROUPS_, 256, 0, stream>>>(x, gns, gnb, h_f);
  cvt(h_f, hb, P);
  cvt(q1w, wqb, sA2); cvt(k1w, wkb, sA2); cvt(v1w, wvb, sA2);
  cvt(projw, wpb, sA2); cvt(postw, wpostb, sA2);

  // 2) q/k/v = dwconv3(conv1x1(h_))
  gemm(wqb, hb, t0f, C_, HW_, C_, C_, HW_, HW_, 0, 0, 1.f, q1b, nullptr, 0, sP, sP);
  dwconv_kernel<<<B_ * C_, 256, 0, stream>>>(t0f, q2w, q2b, qf, C_, H_, W_, 3);
  gemm(wkb, hb, t0f, C_, HW_, C_, C_, HW_, HW_, 0, 0, 1.f, k1b, nullptr, 0, sP, sP);
  dwconv_kernel<<<B_ * C_, 256, 0, stream>>>(t0f, k2w, k2b, kf, C_, H_, W_, 3);
  gemm(wvb, hb, t0f, C_, HW_, C_, C_, HW_, HW_, 0, 0, 1.f, v1b, nullptr, 0, sP, sP);
  dwconv_kernel<<<B_ * C_, 256, 0, stream>>>(t0f, v2w, v2b, vf, C_, H_, W_, 3);
  cvt(qf, qb, P); cvt(kf, kb, P); cvt(vf, vb, P);

  // 3) scores = q^T k * C^-0.5 ; softmax rows ; h2 = v * attn^T
  gemm(qb, kb, scores, HW_, HW_, C_, HW_, HW_, HW_, 1, 0, 0.0625f, nullptr, nullptr, sP, sP, sS);
  softmax_rows_kernel<<<B_ * HW_, 256, 0, stream>>>(scores, scores, HW_);
  cvt(scores, scoresb, PS);
  gemm(vb, scoresb, h2f, C_, HW_, HW_, HW_, HW_, HW_, 0, 1, 1.f, nullptr, nullptr, sP, sS, sP);
  cvt(h2f, hb, P);  // reuse hb as h2 bf16
  gemm(wpb, hb, h2p, C_, HW_, C_, C_, HW_, HW_, 0, 0, 1.f, projb, nullptr, 0, sP, sP);

  // 4) Laplacian channel attention on raw x
  softmax_rows_kernel<<<B_ * C_, 256, 0, stream>>>(x, smf, HW_);
  cvt(x, xb, P); cvt(smf, smb, P);
  gemm(smb, xb, attf, C_, C_, HW_, HW_, HW_, C_, 0, 1, 1.f, nullptr, nullptr, sP, sP, sA2);
  softmax_rows_kernel<<<B_ * C_, 256, 0, stream>>>(attf, attf, C_);
  gauss_dw_kernel<<<B_ * C_, 256, 0, stream>>>(x, G1f, H_, W_, 3, 1.6f);
  gauss_dw_kernel<<<B_ * C_, 256, 0, stream>>>(G1f, G2f, H_, W_, 5, 1.6f * 1.2599210498948732f);
  // L1 = x - G1
  subtract_kernel<<<dim3((unsigned)((P + 255) / 256)), 256, 0, stream>>>(x, G1f, Lf, P);
  softmax_rows_kernel<<<B_ * C_, 256, 0, stream>>>(Lf, smf, HW_);
  cvt(Lf, xb, P); cvt(smf, smb, P);
  gemm(smb, xb, attf, C_, C_, HW_, HW_, HW_, C_, 0, 1, 1.f, nullptr, attf, sP, sP, sA2);
  // L2 = G1 - G2
  subtract_kernel<<<dim3((unsigned)((P + 255) / 256)), 256, 0, stream>>>(G1f, G2f, Lf, P);
  softmax_rows_kernel<<<B_ * C_, 256, 0, stream>>>(Lf, smf, HW_);
  cvt(Lf, xb, P); cvt(smf, smb, P);
  gemm(smb, xb, attf, C_, C_, HW_, HW_, HW_, C_, 0, 1, 1.f, nullptr, attf, sP, sP, sA2);
  // fa = fc^T q
  cvt(attf, attb, CC);
  gemm(attb, qb, faf, C_, HW_, C_, C_, HW_, HW_, 1, 0, 1.f, nullptr, nullptr, sA2, sP, sP);

  // 5) FFT interaction
  const long long rowsHC = (long long)B_ * C_ * H_;
  rfft_w_kernel<<<dim3((unsigned)((rowsHC * WF_ + 255) / 256)), 256, 0, stream>>>(h2p, tre, tim, rowsHC);
  fft_h_kernel<<<dim3((unsigned)((PF + 255) / 256)), 256, 0, stream>>>(tre, tim, FeR, FeI, (long long)B_ * C_, -1.f, 1.f);
  rfft_w_kernel<<<dim3((unsigned)((rowsHC * WF_ + 255) / 256)), 256, 0, stream>>>(faf, tre, tim, rowsHC);
  fft_h_kernel<<<dim3((unsigned)((PF + 255) / 256)), 256, 0, stream>>>(tre, tim, FdR, FdI, (long long)B_ * C_, -1.f, 1.f);
  amp_phase_kernel<<<dim3((unsigned)((PF + 255) / 256)), 256, 0, stream>>>(FeR, FeI, FdR, FdI, ampf, phar, PF);
  dwconv_kernel<<<B_ * C_, 256, 0, stream>>>(phar, midw, midb, phaf, C_, H_, WF_, 3);
  polar_kernel<<<dim3((unsigned)((PF + 255) / 256)), 256, 0, stream>>>(ampf, phaf, pcf, psf, PF);
  cvt(pcf, pcb, PF);
  gemm(wpostb, pcb, realf, C_, FC_, C_, C_, FC_, FC_, 0, 0, 1.f, postb, nullptr, 0, sF, sF);
  dwconv_kernel<<<B_ * C_, 256, 0, stream>>>(psf, c1w, c1b, imagf, C_, H_, WF_, 3);
  fft_h_kernel<<<dim3((unsigned)((PF + 255) / 256)), 256, 0, stream>>>(realf, imagf, tre, tim, (long long)B_ * C_, 1.f, 1.f / (float)H_);
  irfft_w_kernel<<<dim3((unsigned)((rowsHC * W_ + 255) / 256)), 256, 0, stream>>>(tre, tim, recf, rowsHC);

  // 6) y = x + rec; out = 2y - mean_hw(y)
  final_kernel<<<B_ * C_, 256, 0, stream>>>(x, recf, out);
}
